// CnnLSTM_7773890805991
// MI455X (gfx1250) — matrix operand order repaired and checked by reading, not compiled, not run on hardware
//
#include <hip/hip_runtime.h>
#include <hip/hip_bf16.h>

// ---------------------------------------------------------------------------
// CDNA5 (gfx1250) persistent bf16-WMMA 2-layer LSTM.
//   B=256, T=2048, K=64, H=256, G=4H=1024, NUM_PRED=512
//   16 workgroups x 512 threads (16 wave32 waves). Workgroup wg owns batch
//   rows [wg*16, wg*16+16). Wave w owns hidden units j = w*16 + (lane&15),
//   i.e. gate tiles {w, w+16, w+32, w+48} of the 64 N-tiles of G=1024.
//   Weights converted once to bf16 (L2-resident, ~1.65 MB); hidden state
//   exchanged through LDS; cell state lives in registers.
// ---------------------------------------------------------------------------

typedef __bf16 bf16;
typedef __attribute__((ext_vector_type(16))) __bf16 v16bf;
typedef __attribute__((ext_vector_type(8)))  float  v8f;

#define TSEQ  2048
#define KCONV 64
#define HDIM  256
#define GDIM  1024

__device__ __forceinline__ bf16 f2bf(float f) {
  unsigned u = __builtin_bit_cast(unsigned, f);
  u += 0x7FFFu + ((u >> 16) & 1u);              // round-to-nearest-even
  unsigned short s = (unsigned short)(u >> 16);
  return __builtin_bit_cast(bf16, s);
}
__device__ __forceinline__ float sigm(float x) { return 1.f / (1.f + __expf(-x)); }
__device__ __forceinline__ float tanh_f(float x) {
  float e = __expf(2.f * x);
  return (e - 1.f) / (e + 1.f);
}

union AF { v16bf v; float4 f[2]; };

// Accumulate this wave's 4 gate tiles:  acc[t] += A(16 x kdim) * B tile.
//   A: LDS, row-major bf16, row stride astride (elements).
//      A-fragment (16x32 bf16): lane holds row m=lane&15,
//      K range [kb + (lane>>4)*16 .. +15] in v0-7
//      -> two contiguous 16B LDS loads.
//   B: bf16 weights W[col][k] row-major (g = h @ W^T so B[k][n] = W[col_n][k]).
//      B-fragment (32x16 bf16): lane holds col n=lane&15,
//      K range [kb + (lane>>4)*16 .. +15] -> 32 contiguous bytes of one row.
// NOTE: "#pragma unroll 2" is deliberate. Full unrolling makes the compiler
// keep every B-fragment live, exceed the 256-VGPR/wave budget (16 waves/WGP)
// and spill fragments to scratch inside the serial recurrence. Unroll-by-2
// double-buffers the fragments (loads of iteration k+1 overlap the WMMAs of
// iteration k) while staying ~150 VGPRs.
__device__ __forceinline__ void gemm_acc(v8f* acc, const bf16* As, int astride,
                                         const bf16* W, int kdim,
                                         int wv, int n, int hi) {
#pragma unroll 2
  for (int kb = 0; kb < kdim; kb += 32) {
    AF a;
    const bf16* ap = As + n * astride + kb + hi * 16;
    a.f[0] = *(const float4*)(ap);
    a.f[1] = *(const float4*)(ap + 8);
    AF b[4];
#pragma unroll
    for (int t4 = 0; t4 < 4; ++t4) {
      const bf16* bp = W + (size_t)((wv + 16 * t4) * 16 + n) * kdim + kb + hi * 16;
      b[t4].f[0] = *(const float4*)(bp);
      b[t4].f[1] = *(const float4*)(bp + 8);
    }
#pragma unroll
    for (int t4 = 0; t4 < 4; ++t4)
      acc[t4] = __builtin_amdgcn_wmma_f32_16x16x32_bf16(
          false, a.v, false, b[t4].v, (short)0, acc[t4], false, false);
  }
}

// ---------------------------------------------------------------------------
// Prep: fp32 -> bf16 weights into workspace; fuse bias pairs.
// ---------------------------------------------------------------------------
__global__ void __launch_bounds__(256)
prep_weights(const float* __restrict__ Wih0, const float* __restrict__ Whh0,
             const float* __restrict__ Wih1, const float* __restrict__ Whh1,
             const float* __restrict__ bih0, const float* __restrict__ bhh0,
             const float* __restrict__ bih1, const float* __restrict__ bhh1,
             bf16* __restrict__ wbf, float* __restrict__ bias01) {
  const int i = blockIdx.x * 256 + threadIdx.x;
  const int N0 = GDIM * KCONV, N1 = GDIM * HDIM;
  if (i < N0)               wbf[i] = f2bf(Wih0[i]);
  else if (i < N0 + N1)     wbf[i] = f2bf(Whh0[i - N0]);
  else if (i < N0 + 2 * N1) wbf[i] = f2bf(Wih1[i - N0 - N1]);
  else if (i < N0 + 3 * N1) wbf[i] = f2bf(Whh1[i - N0 - 2 * N1]);
  if (i < GDIM)             bias01[i] = bih0[i] + bhh0[i];
  else if (i < 2 * GDIM)    bias01[i] = bih1[i - GDIM] + bhh1[i - GDIM];
}

// ---------------------------------------------------------------------------
// Persistent scan + autoregressive kernel.
// ---------------------------------------------------------------------------
__global__ void __launch_bounds__(512)
lstm_persistent(const float* __restrict__ input,
                const float* __restrict__ conv_w, const float* __restrict__ conv_b,
                const bf16* __restrict__ wbf, const float* __restrict__ bias01,
                const float* __restrict__ lin_w, const float* __restrict__ lin_b,
                const int* __restrict__ npred_p, float* __restrict__ out) {
  __shared__ __align__(16) bf16 xs[16][KCONV];   // x_t tile (bf16)
  __shared__ __align__(16) bf16 h0s[16][HDIM];   // layer0 hidden (bf16)
  __shared__ __align__(16) bf16 h1s[16][HDIM];   // layer1 hidden (bf16)
  __shared__ float cws[KCONV], cbs[KCONV];
  __shared__ float predsum[16];
  __shared__ float lasts[16];

  const int tid  = threadIdx.x;
  const int wv   = tid >> 5;        // wave id, 0..15 (wave32)
  const int lane = tid & 31;
  const int n    = lane & 15;       // WMMA col / A row
  const int hi   = lane >> 4;       // half-wave
  const int b0   = blockIdx.x * 16; // batch tile base

  const bf16* Wih0 = wbf;
  const bf16* Whh0 = wbf + GDIM * KCONV;
  const bf16* Wih1 = wbf + GDIM * KCONV + GDIM * HDIM;
  const bf16* Whh1 = wbf + GDIM * KCONV + 2 * GDIM * HDIM;
  const float* bias0 = bias01;
  const float* bias1 = bias01 + GDIM;

  if (tid < KCONV) { cws[tid] = conv_w[tid]; cbs[tid] = conv_b[tid]; }
  for (int i = tid; i < 16 * HDIM; i += 512) {
    ((bf16*)h0s)[i] = f2bf(0.f);
    ((bf16*)h1s)[i] = f2bf(0.f);
  }

  // Per-lane gate biases: gate tile t4 covers column (wv+16*t4)*16 + n.
  float b0r[4], b1r[4];
#pragma unroll
  for (int t4 = 0; t4 < 4; ++t4) {
    int col = (wv + 16 * t4) * 16 + n;
    b0r[t4] = bias0[col];
    b1r[t4] = bias1[col];
  }
  const float lwv = lin_w[wv * 16 + n];
  const float lb  = lin_b[0];
  const int npred = *npred_p;
  const int total = TSEQ + npred - 1;

  float c0[8] = {0.f, 0.f, 0.f, 0.f, 0.f, 0.f, 0.f, 0.f};
  float c1[8] = {0.f, 0.f, 0.f, 0.f, 0.f, 0.f, 0.f, 0.f};

  const int mx = tid >> 5;          // row (0..15) for x generation
  const int k0 = (tid & 31) * 2;    // 2 cols per thread

  __syncthreads();

  for (int t = 0; t < total; ++t) {
    const bool isOut = (t >= TSEQ - 1);

    // ---- build x_t (16 x 64, bf16) ----
    if (t < TSEQ) {
      // reshape (B,K,T)->(B,T,K): slab t uses conv row (t*64)>>11, cols
      // (t*64)&2047 .. +63 of input -- uniform scalar conv coeffs.
      const int f0   = t * KCONV;
      const int colb = f0 & (TSEQ - 1);
      const int rowc = f0 >> 11;
      const float cwv = cws[rowc], cbv = cbs[rowc];
      const float2 v = *(const float2*)(input + (size_t)(b0 + mx) * TSEQ + colb + k0);
      xs[mx][k0 + 0] = f2bf(fmaxf(v.x * cwv + cbv, 0.f));
      xs[mx][k0 + 1] = f2bf(fmaxf(v.y * cwv + cbv, 0.f));
    } else {
      const float last = lasts[mx];
#pragma unroll
      for (int j = 0; j < 2; ++j) {
        int k = k0 + j;
        xs[mx][k] = f2bf(fmaxf(last * cws[k] + cbs[k], 0.f));
      }
    }
    if (isOut && tid < 16) predsum[tid] = 0.f;
    __syncthreads();

    // ---- layer 0: g = x @ Wih0^T + h0 @ Whh0^T + b ----
    v8f acc[4];
#pragma unroll
    for (int t4 = 0; t4 < 4; ++t4)
#pragma unroll
      for (int r = 0; r < 8; ++r) acc[t4][r] = b0r[t4];
    gemm_acc(acc, &xs[0][0],  KCONV, Wih0, KCONV, wv, n, hi);
    gemm_acc(acc, &h0s[0][0], HDIM,  Whh0, HDIM,  wv, n, hi);

    // acc[0]=i, acc[1]=f, acc[2]=g, acc[3]=o for hidden unit j = wv*16+n,
    // rows m = r + 8*hi (WMMA C/D layout).
    float hn0[8];
#pragma unroll
    for (int r = 0; r < 8; ++r) {
      float ig = sigm(acc[0][r]);
      float fg = sigm(acc[1][r]);
      float gg = tanh_f(acc[2][r]);
      float og = sigm(acc[3][r]);
      c0[r] = fg * c0[r] + ig * gg;
      hn0[r] = og * tanh_f(c0[r]);
    }
    __syncthreads();                       // all reads of old h0s done
#pragma unroll
    for (int r = 0; r < 8; ++r)
      h0s[r + 8 * hi][wv * 16 + n] = f2bf(hn0[r]);
    __syncthreads();

    // ---- layer 1: g = h0 @ Wih1^T + h1 @ Whh1^T + b ----
#pragma unroll
    for (int t4 = 0; t4 < 4; ++t4)
#pragma unroll
      for (int r = 0; r < 8; ++r) acc[t4][r] = b1r[t4];
    gemm_acc(acc, &h0s[0][0], HDIM, Wih1, HDIM, wv, n, hi);
    gemm_acc(acc, &h1s[0][0], HDIM, Whh1, HDIM, wv, n, hi);

    float hn1[8];
#pragma unroll
    for (int r = 0; r < 8; ++r) {
      float ig = sigm(acc[0][r]);
      float fg = sigm(acc[1][r]);
      float gg = tanh_f(acc[2][r]);
      float og = sigm(acc[3][r]);
      c1[r] = fg * c1[r] + ig * gg;
      hn1[r] = og * tanh_f(c1[r]);
    }
    __syncthreads();                       // all reads of old h1s done
#pragma unroll
    for (int r = 0; r < 8; ++r)
      h1s[r + 8 * hi][wv * 16 + n] = f2bf(hn1[r]);

    if (isOut) {
      // p[m] = sum_j h1[m][j]*lin_w[j] + lin_b  via LDS float atomics
#pragma unroll
      for (int r = 0; r < 8; ++r)
        atomicAdd(&predsum[r + 8 * hi], hn1[r] * lwv);
    }
    __syncthreads();

    if (isOut) {
      if (tid < 16) {
        float p = predsum[tid] + lb;
        lasts[tid] = p;                    // feedback for AR conv
        out[(size_t)(b0 + tid) * npred + (t - (TSEQ - 1))] = p;
      }
      __syncthreads();
    }
  }
}

// ---------------------------------------------------------------------------
extern "C" void kernel_launch(void* const* d_in, const int* in_sizes, int n_in,
                              void* d_out, int out_size, void* d_ws, size_t ws_size,
                              hipStream_t stream) {
  (void)in_sizes; (void)n_in; (void)out_size; (void)ws_size;
  const float* input  = (const float*)d_in[0];
  const float* conv_w = (const float*)d_in[1];
  const float* conv_b = (const float*)d_in[2];
  const float* W_ih0  = (const float*)d_in[3];
  const float* W_hh0  = (const float*)d_in[4];
  const float* b_ih0  = (const float*)d_in[5];
  const float* b_hh0  = (const float*)d_in[6];
  const float* W_ih1  = (const float*)d_in[7];
  const float* W_hh1  = (const float*)d_in[8];
  const float* b_ih1  = (const float*)d_in[9];
  const float* b_hh1  = (const float*)d_in[10];
  const float* lin_w  = (const float*)d_in[11];
  const float* lin_b  = (const float*)d_in[12];
  const int*   npred  = (const int*)d_in[13];
  float* out = (float*)d_out;

  const size_t WTOT = (size_t)GDIM * KCONV + 3ull * GDIM * HDIM; // 851968 bf16
  bf16*  wbf    = reinterpret_cast<bf16*>(d_ws);
  float* bias01 = reinterpret_cast<float*>(reinterpret_cast<char*>(d_ws) + WTOT * 2);

  const int prepBlocks = (int)((WTOT + 255) / 256);
  prep_weights<<<prepBlocks, 256, 0, stream>>>(W_ih0, W_hh0, W_ih1, W_hh1,
                                               b_ih0, b_hh0, b_ih1, b_hh1,
                                               wbf, bias01);
  lstm_persistent<<<16, 512, 0, stream>>>(input, conv_w, conv_b, wbf, bias01,
                                          lin_w, lin_b, npred, out);
}
